// EAGNN_56126632624862
// MI455X (gfx1250) — matrix operand order repaired and checked by reading, compiled, not run on hardware
//
#include <hip/hip_runtime.h>
#include <hip/hip_bf16.h>

typedef unsigned short u16;
typedef __attribute__((ext_vector_type(16))) __bf16 v16bf;
typedef __attribute__((ext_vector_type(8)))  float  v8f;

#define GRID_N   1024
#define BATCH    32
#define NAGENT   16
#define NEDGE    4096
#define BN_ROWS  (BATCH*GRID_N)        // 32768
#define BE_EDGES (BATCH*NEDGE)         // 131072
#define HID      64
#define MID      208
#define KPAD     224
#define ANPB     (NAGENT*GRID_N)       // 16384
#define RTOT     (BATCH*ANPB)          // 524288

__device__ __forceinline__ u16 f2bf(float f){
  union { float f; unsigned int u; } cv; cv.f = f;
  unsigned int u = cv.u;
  return (u16)((u + 0x7FFFu + ((u >> 16) & 1u)) >> 16);   // RNE f32->bf16
}

__device__ __forceinline__ float wredsum(float v){
#pragma unroll
  for (int o = 16; o > 0; o >>= 1) v += __shfl_xor(v, o, 32);
  return v;
}

__global__ void k_zero(float* __restrict__ p, int n){
  int t = blockIdx.x*blockDim.x + threadIdx.x;
  if (t < n) p[t] = 0.f;
}

// graph_nodes -> xc[:,0:16] (f32) and xcb[:,0:16]; zero xcb pad cols 208..223
__global__ void k_copy_l0(const float* __restrict__ gn, float* __restrict__ xc, u16* __restrict__ xcb){
  int t = blockIdx.x*256 + threadIdx.x;
  if (t >= BN_ROWS*16) return;
  int r = t >> 4, c = t & 15;
  float v = gn[t];
  xc [(size_t)r*MID  + c] = v;
  xcb[(size_t)r*KPAD + c] = f2bf(v);
  xcb[(size_t)r*KPAD + MID + c] = 0;
}

// segment_sum(x[src], dst): one thread = (edge, 4 features)
__global__ void k_scatter(const int* __restrict__ links, const float* __restrict__ x,
                          int xstride, int xoff, float* __restrict__ agg, int F, int perEShift){
  int t = blockIdx.x*256 + threadIdx.x;
  int perE = 1 << perEShift;               // F/4
  if (t >= BE_EDGES*perE) return;
  int eg = t >> perEShift;
  int c  = (t & (perE - 1)) << 2;
  int b  = eg >> 12;                       // / NEDGE
  int e  = eg & (NEDGE - 1);
  const int* Lb = links + (size_t)b*2*NEDGE;
  int s = Lb[e];
  int d = Lb[NEDGE + e];
  const float4 v = *(const float4*)(x + (size_t)(b*GRID_N + s)*xstride + xoff + c);
  float* dp = agg + (size_t)(b*GRID_N + d)*F + c;
  atomicAdd(dp+0, v.x); atomicAdd(dp+1, v.y); atomicAdd(dp+2, v.z); atomicAdd(dp+3, v.w);
}

// relu(layer_norm(agg @ W + b)) -> xc[:,colOff:colOff+64] (f32 + bf16). One wave per row.
__global__ void k_mlp(const float* __restrict__ agg, int Fin,
                      const float* __restrict__ W, const float* __restrict__ bias,
                      const float* __restrict__ gam, const float* __restrict__ bet,
                      float* __restrict__ xc, u16* __restrict__ xcb, int colOff){
  int lane = threadIdx.x & 31;
  int wave = threadIdx.x >> 5;
  int r = blockIdx.x*8 + wave;
  if (r >= BN_ROWS) return;
  const float* arow = agg + (size_t)r*Fin;
  float a0 = bias[lane], a1 = bias[lane + 32];
  for (int k = 0; k < Fin; ++k){
    float xv = arow[k];
    a0 = fmaf(xv, W[k*64 + lane],      a0);
    a1 = fmaf(xv, W[k*64 + lane + 32], a1);
  }
  float mean = wredsum(a0 + a1) * (1.f/64.f);
  float var  = wredsum(a0*a0 + a1*a1) * (1.f/64.f) - mean*mean;
  float rs   = rsqrtf(var + 1e-5f);
  float y0 = fmaxf((a0 - mean)*rs*gam[lane]      + bet[lane],      0.f);
  float y1 = fmaxf((a1 - mean)*rs*gam[lane + 32] + bet[lane + 32], 0.f);
  xc [(size_t)r*MID  + colOff + lane]      = y0;
  xc [(size_t)r*MID  + colOff + lane + 32] = y1;
  xcb[(size_t)r*KPAD + colOff + lane]      = f2bf(y0);
  xcb[(size_t)r*KPAD + colOff + lane + 32] = f2bf(y1);
}

// wt[n*224+k] = bf16(actor1_W[(rowOff+k)*208 + n]), zero-padded K
__global__ void k_build_wt(const float* __restrict__ A1, u16* __restrict__ wt, int rowOff){
  int n = blockIdx.x;        // output feature 0..207
  int k = threadIdx.x;       // 0..223
  float v = (k < MID) ? A1[(size_t)(rowOff + k)*MID + n] : 0.f;
  wt[(size_t)n*KPAD + k] = f2bf(v);
}

// u = edges_W @ W_c ; cvec = edges_b @ W_c + actor1_b
__global__ void k_u_cvec(const float* __restrict__ A1, const float* __restrict__ a1b,
                         const float* __restrict__ eW, const float* __restrict__ eb,
                         float* __restrict__ u, float* __restrict__ cv){
  int f = threadIdx.x;
  if (f >= MID) return;
  float su = 0.f, sc = 0.f;
  for (int k = 0; k < HID; ++k){
    float w = A1[(size_t)(2*MID + k)*MID + f];
    su = fmaf(eW[k], w, su);
    sc = fmaf(eb[k], w, sc);
  }
  u[f]  = su;
  cv[f] = sc + a1b[f];
}

// gather agent rows of xcb -> xab[512 x 224]
__global__ void k_gather(const int* __restrict__ aloc, const u16* __restrict__ xcb, u16* __restrict__ xab){
  int g = blockIdx.x;        // 0..511 = b*16+i
  int k = threadIdx.x;       // 0..223
  int b = g >> 4;
  int loc = aloc[g];
  xab[(size_t)g*KPAD + k] = xcb[(size_t)(b*GRID_N + loc)*KPAD + k];
}

// ---- WMMA GEMM: C[M x 208] = A[M x 224(bf16)] * Bt[208 x 224(bf16)]^T ----
union FragB { v16bf v; uint4 q[2]; };

__global__ void k_gemm_bf16(const u16* __restrict__ A, const u16* __restrict__ Bt,
                            float* __restrict__ C, int nTilesTotal){
  int lane = threadIdx.x & 31;
  int wave = threadIdx.x >> 5;
  int tile = blockIdx.x*8 + wave;
  if (tile >= nTilesTotal) return;      // wave-uniform; EXEC stays all-ones
  int mT = tile / 13, nT = tile % 13;   // 13 = 208/16 column tiles
  int lr = lane & 15, lh = lane >> 4;
  const u16* arow = A  + (size_t)(mT*16 + lr)*KPAD;
  const u16* brow = Bt + (size_t)(nT*16 + lr)*KPAD;
  v8f acc = {};
#pragma unroll
  for (int kk = 0; kk < 7; ++kk){
    FragB fa, fb;
    int ka = kk*32 + lh*8;              // A 16x32 bf16 layout: 8-elem K runs per lane half
    fa.q[0] = *(const uint4*)(arow + ka);
    fa.q[1] = *(const uint4*)(arow + ka + 16);
    int kb = kk*32 + lh*8;              // B 32x16 via transposed rows: 8-elem K runs per lane half
    fb.q[0] = *(const uint4*)(brow + kb);
    fb.q[1] = *(const uint4*)(brow + kb + 16);
    acc = __builtin_amdgcn_wmma_f32_16x16x32_bf16(false, fa.v, false, fb.v,
                                                  (short)0, acc, false, false);
  }
  int row0 = mT*16 + lh*8;              // C/D: VGPR r -> M = lh*8 + r, N = lr
  int col  = nT*16 + lr;
#pragma unroll
  for (int r = 0; r < 8; ++r)
    C[(size_t)(row0 + r)*MID + col] = acc[r];
}

// per-feature sum / sumsq of virtual h1 over all R rows
__global__ void k_stats(const float* __restrict__ Pn, const float* __restrict__ Pa,
                        const int* __restrict__ aloc, const float* __restrict__ dist,
                        const float* __restrict__ u, const float* __restrict__ cv,
                        float* __restrict__ sums, float* __restrict__ sumsq){
  __shared__ int sloc[NAGENT];
  int f = threadIdx.x;
  int bn0 = blockIdx.x * 32;            // 32 (b,n) rows per block, same b
  int b = bn0 >> 10;
  if (f < NAGENT) sloc[f] = aloc[b*NAGENT + f];
  __syncthreads();
  bool act = (f < MID);
  float uf = act ? u[f]  : 0.f;
  float cf = act ? cv[f] : 0.f;
  float aS = 0.f, aQ = 0.f;
  for (int rr = 0; rr < 32; ++rr){
    int bn = bn0 + rr;
    int n  = bn & 1023;
    float pn = act ? Pn[(size_t)bn*MID + f] : 0.f;
#pragma unroll
    for (int i = 0; i < NAGENT; ++i){
      float d  = dist[(size_t)sloc[i]*GRID_N + n];
      float pa = act ? Pa[(size_t)(b*NAGENT + i)*MID + f] : 0.f;
      float v  = pn + pa + d*uf + cf;
      aS += v;
      aQ  = fmaf(v, v, aQ);
    }
  }
  if (act){ atomicAdd(&sums[f], aS); atomicAdd(&sumsq[f], aQ); }
}

__global__ void k_finalize(const float* __restrict__ sums, const float* __restrict__ sumsq,
                           float* __restrict__ meanv, float* __restrict__ rstdv){
  int f = threadIdx.x;
  if (f >= MID) return;
  const float invR = 1.f / (float)RTOT;
  float m = sums[f] * invR;
  float v = sumsq[f] * invR - m*m;      // biased var, matches jnp.var
  meanv[f] = m;
  rstdv[f] = rsqrtf(v + 1e-5f);
}

// fused: BN-normalize -> relu -> dot actor2 -> mask. One wave per (b,i,n) row.
__global__ void k_logits(const float* __restrict__ Pn, const float* __restrict__ Pa,
                         const int* __restrict__ aloc, const float* __restrict__ dist,
                         const float* __restrict__ u, const float* __restrict__ cv,
                         const float* __restrict__ meanv, const float* __restrict__ rstdv,
                         const float* __restrict__ bng, const float* __restrict__ bnb,
                         const float* __restrict__ w2, const float* __restrict__ b2,
                         const int* __restrict__ mask, float* __restrict__ logits){
  int lane = threadIdx.x & 31;
  int wave = threadIdx.x >> 5;
  int r = blockIdx.x*8 + wave;
  if (r >= RTOT) return;
  int b   = r >> 14;
  int rem = r & (ANPB - 1);
  int i   = rem >> 10;
  int n   = rem & 1023;
  int ga  = b*NAGENT + i;
  int loc = aloc[ga];
  float d = dist[(size_t)loc*GRID_N + n];
  const float* pn = Pn + (size_t)(b*GRID_N + n)*MID;
  const float* pa = Pa + (size_t)ga*MID;
  float acc = 0.f;
  for (int f = lane; f < MID; f += 32){
    float h = pn[f] + pa[f] + d*u[f] + cv[f];
    h = (h - meanv[f]) * rstdv[f] * bng[f] + bnb[f];
    h = fmaxf(h, 0.f);
    acc = fmaf(h, w2[f], acc);
  }
  acc = wredsum(acc);
  if (lane == 0){
    float lg = acc + b2[0];
    if (mask[r] == 0) lg = -1e8f;
    logits[r] = lg;
  }
}

__global__ void k_softmax(const float* __restrict__ logits, float* __restrict__ out){
  __shared__ float sred[256];
  int b = blockIdx.x, t = threadIdx.x;
  const float* L = logits + (size_t)b*ANPB;
  float m = -3.4e38f;
  for (int k = t; k < ANPB; k += 256) m = fmaxf(m, L[k]);
  sred[t] = m; __syncthreads();
  for (int s = 128; s > 0; s >>= 1){ if (t < s) sred[t] = fmaxf(sred[t], sred[t+s]); __syncthreads(); }
  float bm = sred[0]; __syncthreads();
  float se = 0.f;
  for (int k = t; k < ANPB; k += 256) se += __expf(L[k] - bm);
  sred[t] = se; __syncthreads();
  for (int s = 128; s > 0; s >>= 1){ if (t < s) sred[t] += sred[t+s]; __syncthreads(); }
  float inv = 1.f / sred[0];
  for (int k = t; k < ANPB; k += 256) out[(size_t)b*ANPB + k] = __expf(L[k] - bm) * inv;
}

extern "C" void kernel_launch(void* const* d_in, const int* in_sizes, int n_in,
                              void* d_out, int out_size, void* d_ws, size_t ws_size,
                              hipStream_t stream){
  (void)in_sizes; (void)n_in; (void)out_size; (void)ws_size;
  const float* gn    = (const float*)d_in[0];
  const int*   links = (const int*)d_in[1];
  const int*   aloc  = (const int*)d_in[2];
  const int*   mask  = (const int*)d_in[3];
  const float* dist  = (const float*)d_in[4];
  const float* cW[3]  = {(const float*)d_in[5], (const float*)d_in[9],  (const float*)d_in[13]};
  const float* cB[3]  = {(const float*)d_in[6], (const float*)d_in[10], (const float*)d_in[14]};
  const float* cG[3]  = {(const float*)d_in[7], (const float*)d_in[11], (const float*)d_in[15]};
  const float* cBe[3] = {(const float*)d_in[8], (const float*)d_in[12], (const float*)d_in[16]};
  const float* eW  = (const float*)d_in[17];
  const float* eb  = (const float*)d_in[18];
  const float* A1  = (const float*)d_in[19];
  const float* a1b = (const float*)d_in[20];
  const float* bng = (const float*)d_in[21];
  const float* bnb = (const float*)d_in[22];
  const float* w2  = (const float*)d_in[23];
  const float* b2  = (const float*)d_in[24];
  float* out = (float*)d_out;

  size_t off = 0;
  auto carve = [&](size_t bytes) -> void* {
    void* p = (char*)d_ws + off;
    off += (bytes + 255) & ~(size_t)255;
    return p;
  };
  float* xc     = (float*)carve((size_t)BN_ROWS*MID*4);
  u16*   xcb    = (u16*)  carve((size_t)BN_ROWS*KPAD*2);
  float* agg    = (float*)carve((size_t)BN_ROWS*HID*4);
  float* Pn     = (float*)carve((size_t)BN_ROWS*MID*4);
  float* Pa     = (float*)carve((size_t)512*MID*4);
  u16*   xab    = (u16*)  carve((size_t)512*KPAD*2);
  u16*   WaT    = (u16*)  carve((size_t)MID*KPAD*2);
  u16*   WbT    = (u16*)  carve((size_t)MID*KPAD*2);
  float* uvec   = (float*)carve(1024);
  float* cvec   = (float*)carve(1024);
  float* sums   = (float*)carve(1024);
  float* sumsq  = (float*)carve(1024);
  float* meanv  = (float*)carve(1024);
  float* rstdv  = (float*)carve(1024);
  float* logits = (float*)carve((size_t)RTOT*4);

  // --- graph convs ---
  k_copy_l0<<<2048, 256, 0, stream>>>(gn, xc, xcb);

  k_zero<<<(BN_ROWS*16 + 255)/256, 256, 0, stream>>>(agg, BN_ROWS*16);
  k_scatter<<<2048, 256, 0, stream>>>(links, gn, 16, 0, agg, 16, 2);
  k_mlp<<<4096, 256, 0, stream>>>(agg, 16, cW[0], cB[0], cG[0], cBe[0], xc, xcb, 16);

  k_zero<<<(BN_ROWS*64 + 255)/256, 256, 0, stream>>>(agg, BN_ROWS*64);
  k_scatter<<<8192, 256, 0, stream>>>(links, xc, MID, 16, agg, 64, 4);
  k_mlp<<<4096, 256, 0, stream>>>(agg, 64, cW[1], cB[1], cG[1], cBe[1], xc, xcb, 80);

  k_zero<<<(BN_ROWS*64 + 255)/256, 256, 0, stream>>>(agg, BN_ROWS*64);
  k_scatter<<<8192, 256, 0, stream>>>(links, xc, MID, 80, agg, 64, 4);
  k_mlp<<<4096, 256, 0, stream>>>(agg, 64, cW[2], cB[2], cG[2], cBe[2], xc, xcb, 144);

  // --- actor1 decomposition ---
  k_build_wt<<<MID, KPAD, 0, stream>>>(A1, WaT, 0);
  k_build_wt<<<MID, KPAD, 0, stream>>>(A1, WbT, 208);
  k_u_cvec<<<1, 256, 0, stream>>>(A1, a1b, eW, eb, uvec, cvec);

  k_gemm_bf16<<<3328, 256, 0, stream>>>(xcb, WbT, Pn, 2048*13);   // P_n: 32768x208
  k_gather<<<512, KPAD, 0, stream>>>(aloc, xcb, xab);
  k_gemm_bf16<<<52, 256, 0, stream>>>(xab, WaT, Pa, 32*13);       // P_a: 512x208

  // --- batchnorm stats over virtual h1 ---
  k_zero<<<1, 256, 0, stream>>>(sums, MID);
  k_zero<<<1, 256, 0, stream>>>(sumsq, MID);
  k_stats<<<1024, 256, 0, stream>>>(Pn, Pa, aloc, dist, uvec, cvec, sums, sumsq);
  k_finalize<<<1, 256, 0, stream>>>(sums, sumsq, meanv, rstdv);

  // --- fused BN/relu/actor2/mask, then softmax ---
  k_logits<<<65536, 256, 0, stream>>>(Pn, Pa, aloc, dist, uvec, cvec,
                                      meanv, rstdv, bng, bnb, w2, b2, mask, logits);
  k_softmax<<<BATCH, 256, 0, stream>>>(logits, out);
}
